// Attend_1846835938146
// MI455X (gfx1250) — matrix operand order repaired and checked by reading, not compiled, not run on hardware
//
#include <hip/hip_runtime.h>
#include <float.h>
#include <math.h>

typedef __attribute__((ext_vector_type(16))) _Float16 v16h;
typedef __attribute__((ext_vector_type(8)))  float    v8f;

#define ATT_B 4
#define ATT_H 8
#define ATT_N 2048
#define ATT_D 64
#define ATT_BH (ATT_B * ATT_H)

#define WAVES_PER_WG 4
#define ROWS_PER_WAVE 16
#define ROWS_PER_WG (WAVES_PER_WG * ROWS_PER_WAVE)

#define LOG2E 1.44269504088896340736f
#define MASK_BIG 3.0e38f

// ---------------- workspace layout ----------------
// kfrag: f16 K in WMMA-B fragment order:
//   record[((bh*(N/16) + jt)*2 + c)*32 + lane] = 16 halves,
//   element i = K[jt*16 + (lane&15)][c*32 + (lane>>4)*16 + i]
// vfrag: f16 V in WMMA-B fragment order:
//   record[((bh*(N/32) + j32)*4 + dn)*32 + lane] = 16 halves,
//   element i = V[j32*32 + (lane>>4)*16 + i][dn*16 + (lane&15)]
// kbias: float per key = mask ? |k|^2 * log2(e) : MASK_BIG
#define KFRAG_HALVES ((size_t)ATT_BH * (ATT_N / 16) * 2 * 32 * 16)
#define VFRAG_HALVES ((size_t)ATT_BH * (ATT_N / 32) * 4 * 32 * 16)
#define KBIAS_FLOATS ((size_t)ATT_BH * ATT_N)

// ================= pre-pass kernels =================

__global__ __launch_bounds__(256)
void pack_k(const float* __restrict__ K, _Float16* __restrict__ kf)
{
    size_t t = (size_t)blockIdx.x * 256 + threadIdx.x;   // 524288 threads, 8 halves each
    int g    = (int)( t        & 1);
    int lane = (int)((t >> 1)  & 31);
    int c    = (int)((t >> 6)  & 1);
    int jt   = (int)((t >> 7)  & 127);
    int bh   = (int)( t >> 14);
    int row   = jt * 16 + (lane & 15);
    int depth = c * 32 + (lane >> 4) * 16 + g * 8;
    const float* src = K + ((size_t)bh * ATT_N + row) * ATT_D + depth;
    _Float16* dst = kf + t * 8;
    #pragma unroll
    for (int s = 0; s < 8; ++s) dst[s] = (_Float16)src[s];
}

__global__ __launch_bounds__(256)
void pack_v(const float* __restrict__ V, _Float16* __restrict__ vf)
{
    size_t t = (size_t)blockIdx.x * 256 + threadIdx.x;   // 524288 threads, 8 halves each
    int g    = (int)( t        & 1);
    int lane = (int)((t >> 1)  & 31);
    int dn   = (int)((t >> 6)  & 3);
    int j32  = (int)((t >> 8)  & 63);
    int bh   = (int)( t >> 14);
    int col  = dn * 16 + (lane & 15);
    int row0 = j32 * 32 + (lane >> 4) * 16 + g * 8;
    const float* src = V + ((size_t)bh * ATT_N + row0) * ATT_D + col;
    _Float16* dst = vf + t * 8;
    #pragma unroll
    for (int s = 0; s < 8; ++s) dst[s] = (_Float16)src[(size_t)s * ATT_D];
}

__global__ __launch_bounds__(256)
void pack_kbias(const float* __restrict__ K, const unsigned char* __restrict__ M,
                float* __restrict__ kb)
{
    size_t t = (size_t)blockIdx.x * 256 + threadIdx.x;   // BH*N = 65536 threads
    int j  = (int)(t & (ATT_N - 1));
    int bh = (int)(t >> 11);
    int b  = bh / ATT_H;
    const float* src = K + ((size_t)bh * ATT_N + j) * ATT_D;
    float s = 0.f;
    #pragma unroll
    for (int i = 0; i < ATT_D; ++i) { float x = src[i]; s += x * x; }
    kb[t] = M[(size_t)b * ATT_N + j] ? (s * LOG2E) : MASK_BIG;
}

// ================= main flash-attention kernel (pre-packed path) =================
// sim(log2 domain) = (0.25*log2e)*(q.k) - log2e*|q|^2 - kbias ; softmax via exp2.
// 64 keys per loop iteration: 16 WMMAs, one P-transpose barrier pair, one
// cross-lane reduction set per row.
__global__ __launch_bounds__(32 * WAVES_PER_WG)
void attend_main(const float* __restrict__ Q,
                 const v16h* __restrict__ KF, const v16h* __restrict__ VF,
                 const float* __restrict__ KB, float* __restrict__ O)
{
    __shared__ float    s_qsq[WAVES_PER_WG][16];
    __shared__ _Float16 s_p[WAVES_PER_WG][16][72];   // 16x64 P tile, rows 144B (16B aligned)

    const int wave = threadIdx.x >> 5;
    const int lane = threadIdx.x & 31;
    const int half = lane >> 4;
    const int l16  = lane & 15;

    const int bh = blockIdx.y;
    const int i0 = blockIdx.x * ROWS_PER_WG + wave * ROWS_PER_WAVE;

    const float* qb   = Q  + ((size_t)bh * ATT_N + i0) * ATT_D;
    const v16h*  kfbh = KF + (size_t)bh * (ATT_N / 16) * 2 * 32;
    const v16h*  vfbh = VF + (size_t)bh * (ATT_N / 32) * 4 * 32;
    const float* kbbh = KB + (size_t)bh * ATT_N;

    // ---- Q tile -> two A fragments, pre-scaled by 0.25*log2(e); |q|^2 via LDS ----
    v16h qa0, qa1;
    {
        const float qscale = 0.25f * LOG2E;
        const float* qp = qb + (size_t)l16 * ATT_D;
        #pragma unroll
        for (int t = 0; t < 8; ++t) {
            qa0[t]     = (_Float16)(qscale * qp[ 0 + half * 16 + t]);
            qa0[t + 8] = (_Float16)(qscale * qp[ 8 + half * 16 + t]);
            qa1[t]     = (_Float16)(qscale * qp[32 + half * 16 + t]);
            qa1[t + 8] = (_Float16)(qscale * qp[40 + half * 16 + t]);
        }
        float ps = 0.f;
        const float* qq = qp + half * 32;
        #pragma unroll
        for (int t = 0; t < 32; ++t) { float x = qq[t]; ps += x * x; }
        ps += __shfl_xor(ps, 16);
        s_qsq[wave][l16] = ps;
    }
    __syncthreads();
    float qsq[8];
    #pragma unroll
    for (int r = 0; r < 8; ++r) qsq[r] = s_qsq[wave][r + 8 * half] * LOG2E;

    float m_i[8], l_i[8];
    v8f oacc[4];
    {
        v8f z = {};
        #pragma unroll
        for (int dn = 0; dn < 4; ++dn) oacc[dn] = z;
        #pragma unroll
        for (int r = 0; r < 8; ++r) { m_i[r] = -FLT_MAX; l_i[r] = 0.f; }
    }

    for (int j0 = 0; j0 < ATT_N; j0 += 64) {
        const int jt  = j0 >> 4;   // four 16-key score subtiles
        const int j32 = j0 >> 5;   // two 32-key PV tiles
        if (j0 + 64 < ATT_N) {   // global_prefetch_b8 of next fragment records
            __builtin_prefetch(kfbh + ((size_t)(jt + 4) * 2) * 32 + lane, 0, 0);
            __builtin_prefetch(vfbh + ((size_t)(j32 + 2) * 4) * 32 + lane, 0, 0);
        }

        // ---- S = Q.K^T : 8 WMMAs, fragments pre-packed (2x b128 loads each) ----
        v8f sarr[4];
        #pragma unroll
        for (int sub = 0; sub < 4; ++sub) {
            v16h kf0 = kfbh[((size_t)(jt + sub) * 2 + 0) * 32 + lane];
            v16h kf1 = kfbh[((size_t)(jt + sub) * 2 + 1) * 32 + lane];
            v8f sa = {};
            sa = __builtin_amdgcn_wmma_f32_16x16x32_f16(false, qa0, false, kf0, (short)0, sa, false, false);
            sa = __builtin_amdgcn_wmma_f32_16x16x32_f16(false, qa1, false, kf1, (short)0, sa, false, false);
            sarr[sub] = sa;
        }
        float kbv[4];
        #pragma unroll
        for (int sub = 0; sub < 4; ++sub) kbv[sub] = kbbh[j0 + sub * 16 + l16];

        // ---- online softmax (base-2) over 64 keys; row = r+8*half, col = sub*16+l16 ----
        #pragma unroll
        for (int r = 0; r < 8; ++r) {
            float a[4];
            #pragma unroll
            for (int sub = 0; sub < 4; ++sub) a[sub] = sarr[sub][r] - qsq[r] - kbv[sub];
            float t = fmaxf(fmaxf(a[0], a[1]), fmaxf(a[2], a[3]));
            t = fmaxf(t, __shfl_xor(t, 1));
            t = fmaxf(t, __shfl_xor(t, 2));
            t = fmaxf(t, __shfl_xor(t, 4));
            t = fmaxf(t, __shfl_xor(t, 8));
            float mnew = fmaxf(m_i[r], t);
            float e[4];
            #pragma unroll
            for (int sub = 0; sub < 4; ++sub) e[sub] = exp2f(a[sub] - mnew);  // v_exp_f32
            float rs = (e[0] + e[1]) + (e[2] + e[3]);
            rs += __shfl_xor(rs, 1);
            rs += __shfl_xor(rs, 2);
            rs += __shfl_xor(rs, 4);
            rs += __shfl_xor(rs, 8);
            float corr = exp2f(m_i[r] - mnew);
            l_i[r] = l_i[r] * corr + rs;
            m_i[r] = mnew;
            #pragma unroll
            for (int dn = 0; dn < 4; ++dn) oacc[dn][r] *= corr;
            #pragma unroll
            for (int sub = 0; sub < 4; ++sub)
                s_p[wave][r + 8 * half][sub * 16 + l16] = (_Float16)e[sub];
        }
        __syncthreads();

        // ---- read P back as two A fragments: 2x ds_load_b128 each ----
        v16h pa0, pa1;
        {
            union { uint4 u[2]; v16h h; } pu;
            const uint4* prow = (const uint4*)&s_p[wave][l16][0];
            pu.u[0] = prow[2 * half];     pu.u[1] = prow[2 * half + 1]; pa0 = pu.h; // keys 0..31
            pu.u[0] = prow[4 + 2 * half]; pu.u[1] = prow[5 + 2 * half]; pa1 = pu.h; // keys 32..63
        }
        __syncthreads();

        // ---- O += P.V : 8 WMMAs, V fragments pre-packed (2x b128 loads each) ----
        #pragma unroll
        for (int dn = 0; dn < 4; ++dn) {
            v16h v0 = vfbh[((size_t)(j32 + 0) * 4 + dn) * 32 + lane];
            v16h v1 = vfbh[((size_t)(j32 + 1) * 4 + dn) * 32 + lane];
            oacc[dn] = __builtin_amdgcn_wmma_f32_16x16x32_f16(false, pa0, false, v0, (short)0, oacc[dn], false, false);
            oacc[dn] = __builtin_amdgcn_wmma_f32_16x16x32_f16(false, pa1, false, v1, (short)0, oacc[dn], false, false);
        }
    }

    // ---- normalize and store ----
    float* ob = O + ((size_t)bh * ATT_N + i0) * ATT_D;
    #pragma unroll
    for (int r = 0; r < 8; ++r) {
        float inv = 1.0f / l_i[r];
        #pragma unroll
        for (int dn = 0; dn < 4; ++dn)
            ob[(size_t)(r + 8 * half) * ATT_D + dn * 16 + l16] = oacc[dn][r] * inv;
    }
}

// ================= fallback: self-contained kernel (no workspace) =================
__global__ __launch_bounds__(32 * WAVES_PER_WG)
void attend_fb(const float* __restrict__ Q, const float* __restrict__ K,
               const float* __restrict__ V, const unsigned char* __restrict__ M,
               float* __restrict__ O)
{
    __shared__ float s_qsq[WAVES_PER_WG][16];
    __shared__ float s_p[WAVES_PER_WG][16][33];

    const int wave = threadIdx.x >> 5;
    const int lane = threadIdx.x & 31;
    const int half = lane >> 4;
    const int l16  = lane & 15;

    const int bh = blockIdx.y;
    const int b  = bh / ATT_H;
    const int i0 = blockIdx.x * ROWS_PER_WG + wave * ROWS_PER_WAVE;

    const float* qb = Q + ((size_t)bh * ATT_N + i0) * ATT_D;
    const float* kb = K + (size_t)bh * ATT_N * ATT_D;
    const float* vb = V + (size_t)bh * ATT_N * ATT_D;
    const unsigned char* mb = M + (size_t)b * ATT_N;

    v16h qa0, qa1;
    {
        const float* qp = qb + (size_t)l16 * ATT_D;
        #pragma unroll
        for (int t = 0; t < 8; ++t) {
            qa0[t]     = (_Float16)qp[ 0 + half * 16 + t];
            qa0[t + 8] = (_Float16)qp[ 8 + half * 16 + t];
            qa1[t]     = (_Float16)qp[32 + half * 16 + t];
            qa1[t + 8] = (_Float16)qp[40 + half * 16 + t];
        }
        float ps = 0.f;
        const float* qq = qp + half * 32;
        #pragma unroll
        for (int t = 0; t < 32; ++t) { float x = qq[t]; ps += x * x; }
        ps += __shfl_xor(ps, 16);
        s_qsq[wave][l16] = ps;
    }
    __syncthreads();
    float qsq[8];
    #pragma unroll
    for (int r = 0; r < 8; ++r) qsq[r] = s_qsq[wave][r + 8 * half];

    float m_i[8], l_i[8];
    v8f oacc[4];
    {
        v8f z = {};
        #pragma unroll
        for (int dn = 0; dn < 4; ++dn) oacc[dn] = z;
        #pragma unroll
        for (int r = 0; r < 8; ++r) { m_i[r] = -FLT_MAX; l_i[r] = 0.f; }
    }

    for (int j0 = 0; j0 < ATT_N; j0 += 32) {
        v8f   sarr[2];
        float ksq[2];
        #pragma unroll
        for (int sub = 0; sub < 2; ++sub) {
            const int jbase = j0 + sub * 16;
            const float* kp = kb + (size_t)(jbase + l16) * ATT_D + half * 16;
            v16h kf0, kf1;
            float kq = 0.f;
            #pragma unroll
            for (int i = 0; i < 16; ++i) { float x = kp[i];      kf0[i] = (_Float16)x; kq += x * x; }
            #pragma unroll
            for (int i = 0; i < 16; ++i) { float x = kp[32 + i]; kf1[i] = (_Float16)x; kq += x * x; }
            kq += __shfl_xor(kq, 16);
            ksq[sub] = kq;
            v8f sa = {};
            sa = __builtin_amdgcn_wmma_f32_16x16x32_f16(false, qa0, false, kf0, (short)0, sa, false, false);
            sa = __builtin_amdgcn_wmma_f32_16x16x32_f16(false, qa1, false, kf1, (short)0, sa, false, false);
            sarr[sub] = sa;
        }
        const bool ok0 = mb[j0 + l16]      != 0;
        const bool ok1 = mb[j0 + 16 + l16] != 0;

        float p0[8], p1[8];
        #pragma unroll
        for (int r = 0; r < 8; ++r) {
            float a = ok0 ? (0.25f * sarr[0][r] - qsq[r] - ksq[0]) : -FLT_MAX;
            float c = ok1 ? (0.25f * sarr[1][r] - qsq[r] - ksq[1]) : -FLT_MAX;
            float t = fmaxf(a, c);
            t = fmaxf(t, __shfl_xor(t, 1));
            t = fmaxf(t, __shfl_xor(t, 2));
            t = fmaxf(t, __shfl_xor(t, 4));
            t = fmaxf(t, __shfl_xor(t, 8));
            float mnew = fmaxf(m_i[r], t);
            float e0 = __expf(a - mnew);
            float e1 = __expf(c - mnew);
            float rs = e0 + e1;
            rs += __shfl_xor(rs, 1);
            rs += __shfl_xor(rs, 2);
            rs += __shfl_xor(rs, 4);
            rs += __shfl_xor(rs, 8);
            float corr = __expf(m_i[r] - mnew);
            l_i[r] = l_i[r] * corr + rs;
            m_i[r] = mnew;
            #pragma unroll
            for (int dn = 0; dn < 4; ++dn) oacc[dn][r] *= corr;
            p0[r] = e0; p1[r] = e1;
        }

        #pragma unroll
        for (int r = 0; r < 8; ++r) {
            s_p[wave][r + 8 * half][l16]      = p0[r];
            s_p[wave][r + 8 * half][16 + l16] = p1[r];
        }
        __syncthreads();
        v16h pa;
        #pragma unroll
        for (int t = 0; t < 8; ++t) {
            pa[t]     = (_Float16)s_p[wave][l16][half * 16 + t];
            pa[t + 8] = (_Float16)s_p[wave][l16][half * 16 + 8 + t];
        }
        __syncthreads();

        const float* vp0 = vb + (size_t)(j0 + half * 16) * ATT_D + l16;
        #pragma unroll
        for (int dn = 0; dn < 4; ++dn) {
            v16h vf;
            const float* vp = vp0 + dn * 16;
            #pragma unroll
            for (int i = 0; i < 16; ++i) vf[i] = (_Float16)vp[(size_t)i * ATT_D];
            oacc[dn] = __builtin_amdgcn_wmma_f32_16x16x32_f16(false, pa, false, vf, (short)0, oacc[dn], false, false);
        }
    }

    float* ob = O + ((size_t)bh * ATT_N + i0) * ATT_D;
    #pragma unroll
    for (int r = 0; r < 8; ++r) {
        float inv = 1.0f / l_i[r];
        #pragma unroll
        for (int dn = 0; dn < 4; ++dn)
            ob[(size_t)(r + 8 * half) * ATT_D + dn * 16 + l16] = oacc[dn][r] * inv;
    }
}

// ================= host launch =================
extern "C" void kernel_launch(void* const* d_in, const int* in_sizes, int n_in,
                              void* d_out, int out_size, void* d_ws, size_t ws_size,
                              hipStream_t stream) {
    (void)in_sizes; (void)n_in; (void)out_size;
    const float* q = (const float*)d_in[0];
    const float* k = (const float*)d_in[1];
    const float* v = (const float*)d_in[2];
    const unsigned char* m = (const unsigned char*)d_in[3];   // jnp bool_ = 1 byte/elem
    float* out = (float*)d_out;

    const size_t kBytes = KFRAG_HALVES * sizeof(_Float16);   // 8 MB
    const size_t vBytes = VFRAG_HALVES * sizeof(_Float16);   // 8 MB
    const size_t bBytes = KBIAS_FLOATS * sizeof(float);      // 256 KB
    const size_t need   = kBytes + vBytes + bBytes;

    dim3 grid(ATT_N / ROWS_PER_WG, ATT_BH);   // (32, 32)
    dim3 block(32 * WAVES_PER_WG);            // 128 threads = 4 waves

    if (d_ws != nullptr && ws_size >= need) {
        _Float16* kf = (_Float16*)d_ws;
        _Float16* vf = (_Float16*)((char*)d_ws + kBytes);
        float*    kb = (float*)   ((char*)d_ws + kBytes + vBytes);
        hipLaunchKernelGGL(pack_k,     dim3(2048), dim3(256), 0, stream, k, kf);
        hipLaunchKernelGGL(pack_v,     dim3(2048), dim3(256), 0, stream, v, vf);
        hipLaunchKernelGGL(pack_kbias, dim3(256),  dim3(256), 0, stream, k, m, kb);
        hipLaunchKernelGGL(attend_main, grid, block, 0, stream,
                           q, (const v16h*)kf, (const v16h*)vf, (const float*)kb, out);
    } else {
        hipLaunchKernelGGL(attend_fb, grid, block, 0, stream, q, k, v, m, out);
    }
}
